// EGNNLayer_7275674599753
// MI455X (gfx1250) — matrix operand order repaired and checked by reading, not compiled, not run on hardware
//
#include <hip/hip_runtime.h>
#include <hip/hip_bf16.h>

typedef __attribute__((ext_vector_type(16))) __bf16        v16bf;
typedef __attribute__((ext_vector_type(8)))  float         v8f;
typedef __attribute__((ext_vector_type(8)))  unsigned int  v8u;

#define NNODES 50000
#define DIM 128

__device__ __forceinline__ unsigned short f2bf(float f) {
    unsigned u = __float_as_uint(f);
    unsigned r = u + 0x7FFFu + ((u >> 16) & 1u);   // round-to-nearest-even
    return (unsigned short)(r >> 16);
}
__device__ __forceinline__ float silu(float x) { return x / (1.f + __expf(-x)); }
__device__ __forceinline__ v8f vzero() { v8f z = {0.f,0.f,0.f,0.f,0.f,0.f,0.f,0.f}; return z; }

__device__ __forceinline__ v8f wmma_bf16(v8u a, v8u b, v8f c) {
    return __builtin_amdgcn_wmma_f32_16x16x32_bf16(
        false, __builtin_bit_cast(v16bf, a),
        false, __builtin_bit_cast(v16bf, b),
        (short)0, c, false, false);
}

// A-fragment (16x32 bf16, MxK) gathered from LDS, row stride `rs` halfs.
// ISA layout: lanes0-15 M=0..15 hold K pairs {0..7,16..23}; lanes16-31 hold {8..15,24..31}.
__device__ __forceinline__ v8u load_a(const unsigned short* lds, int rs, int kc, int lane) {
    int m  = lane & 15;
    int kb = (lane >> 4) * 8;
    const unsigned short* p = lds + m * rs + kc * 32;
    v8u a;
#pragma unroll
    for (int pr = 0; pr < 8; ++pr) {
        int k0 = (pr >> 2) * 16 + kb + (pr & 3) * 2;
        a[pr] = *(const unsigned int*)(p + k0);
    }
    return a;
}

// Pre-swizzle fp32 weight [K][128] (K = 32*KC) into bf16 B-fragments:
// frag index ((nt*KC + kc)*32 + lane), 16 halfs each.
// B layout: col n = nt*16 + lane%16; k = kc*32 + (e/8)*16 + (lane>=16?8:0) + e%8.
__global__ void swizzle_w(const float* __restrict__ w, unsigned short* __restrict__ out, int KC) {
    int s = blockIdx.x * 256 + threadIdx.x;
    if (s >= 8 * KC * 32 * 16) return;
    int e    = s & 15;
    int lane = (s >> 4) & 31;
    int kc   = (s >> 9) % KC;
    int nt   = (s >> 9) / KC;
    int k = kc * 32 + (e >> 3) * 16 + ((lane >= 16) ? 8 : 0) + (e & 7);
    int n = nt * 16 + (lane & 15);
    out[s] = f2bf(w[k * 128 + n]);
}

__global__ void init_kernel(const float* __restrict__ x, float* __restrict__ agg_h,
                            float* __restrict__ x_out) {
    int i = blockIdx.x * 256 + threadIdx.x;
    if (i < NNODES * DIM) agg_h[i] = 0.f;
    if (i < NNODES * 3)   x_out[i] = x[i];
}

// 32 edges per block; 8 waves; wave wv owns N-tile wv (16 cols) and both M-tiles.
__global__ void __launch_bounds__(256) edge_kernel(
    const float* __restrict__ h, const float* __restrict__ x, const int* __restrict__ ei,
    int E,
    const float* __restrict__ ew1,  // fp32 [257][128], row 256 used here
    const float* __restrict__ eb1, const float* __restrict__ eb2,
    const float* __restrict__ cb1, const float* __restrict__ cw2,
    const unsigned short* __restrict__ we1s, const unsigned short* __restrict__ we2s,
    const unsigned short* __restrict__ wc1s,
    float* __restrict__ agg_h, float* __restrict__ x_out)
{
    __shared__ unsigned short ein[32][256];  // bf16 edge_input (first 256 cols)  16KB
    __shared__ unsigned short h1[32][128];   // bf16 silu(layer1)                  8KB
    __shared__ unsigned short mb[32][128];   // bf16 m                             8KB
    __shared__ float tb[32][128];            // fp32 silu(coord layer1)           16KB
    __shared__ int rows[32], cols[32];
    __shared__ float cdiff[32][3];
    __shared__ float radial[32];
    __shared__ float parts[32][8];

    int tid = threadIdx.x;
    int ebase = blockIdx.x * 32;

    if (tid < 32) {
        int e = ebase + tid;
        int r = ei[e], c = ei[E + e];
        rows[tid] = r; cols[tid] = c;
        float rad = 0.f;
#pragma unroll
        for (int d = 0; d < 3; ++d) {
            float dd = x[r * 3 + d] - x[c * 3 + d];
            cdiff[tid][d] = dd;
            rad += dd * dd;
        }
        radial[tid] = rad;
    }
    __syncthreads();

    for (int i = tid; i < 32 * 256; i += 256) {
        int m = i >> 8, k = i & 255;
        float v = (k < 128) ? h[(size_t)rows[m] * 128 + k] : h[(size_t)cols[m] * 128 + (k - 128)];
        ein[m][k] = f2bf(v);
    }
    __syncthreads();

    int wv = tid >> 5, lane = tid & 31;
    int ncol = wv * 16 + (lane & 15);
    int mrow0 = (lane >> 4) * 8;

    // ---- edge layer1: [32x257] @ [257x128] ----
    {
        v8f acc0 = vzero(), acc1 = vzero();
#pragma unroll
        for (int kc = 0; kc < 8; ++kc) {
            v8u b  = *(const v8u*)(we1s + (((wv * 8 + kc) * 32 + lane) << 4));
            v8u a0 = load_a(&ein[0][0],  256, kc, lane);
            v8u a1 = load_a(&ein[16][0], 256, kc, lane);
            acc0 = wmma_bf16(a0, b, acc0);
            acc1 = wmma_bf16(a1, b, acc1);
        }
        float w256 = ew1[256 * 128 + ncol];  // radial column (K=256), fp32 rank-1 update
        float bias = eb1[ncol];
#pragma unroll
        for (int r2 = 0; r2 < 8; ++r2) {
            int m0 = mrow0 + r2;
            h1[m0][ncol]      = f2bf(silu(acc0[r2] + radial[m0]      * w256 + bias));
            h1[m0 + 16][ncol] = f2bf(silu(acc1[r2] + radial[m0 + 16] * w256 + bias));
        }
    }
    __syncthreads();

    // ---- edge layer2: m = h1 @ e_w2 + b2 ----
    {
        v8f acc0 = vzero(), acc1 = vzero();
#pragma unroll
        for (int kc = 0; kc < 4; ++kc) {
            v8u b  = *(const v8u*)(we2s + (((wv * 4 + kc) * 32 + lane) << 4));
            v8u a0 = load_a(&h1[0][0],  128, kc, lane);
            v8u a1 = load_a(&h1[16][0], 128, kc, lane);
            acc0 = wmma_bf16(a0, b, acc0);
            acc1 = wmma_bf16(a1, b, acc1);
        }
        float bias = eb2[ncol];
#pragma unroll
        for (int r2 = 0; r2 < 8; ++r2) {
            int m0 = mrow0 + r2;
            float v0 = acc0[r2] + bias;
            float v1 = acc1[r2] + bias;
            mb[m0][ncol]      = f2bf(v0);
            mb[m0 + 16][ncol] = f2bf(v1);
            atomicAdd(&agg_h[(size_t)rows[m0]      * 128 + ncol], v0);
            atomicAdd(&agg_h[(size_t)rows[m0 + 16] * 128 + ncol], v1);
        }
    }
    __syncthreads();

    // ---- coord layer1: silu(m @ c_w1 + c_b1) ----
    {
        v8f acc0 = vzero(), acc1 = vzero();
#pragma unroll
        for (int kc = 0; kc < 4; ++kc) {
            v8u b  = *(const v8u*)(wc1s + (((wv * 4 + kc) * 32 + lane) << 4));
            v8u a0 = load_a(&mb[0][0],  128, kc, lane);
            v8u a1 = load_a(&mb[16][0], 128, kc, lane);
            acc0 = wmma_bf16(a0, b, acc0);
            acc1 = wmma_bf16(a1, b, acc1);
        }
        float bias = cb1[ncol];
#pragma unroll
        for (int r2 = 0; r2 < 8; ++r2) {
            int m0 = mrow0 + r2;
            tb[m0][ncol]      = silu(acc0[r2] + bias);
            tb[m0 + 16][ncol] = silu(acc1[r2] + bias);
        }
    }
    __syncthreads();

    // ---- coef = tb . c_w2 (128 -> 1), then scatter trans ----
    {
        int m = tid >> 3, p = tid & 7;   // 32 edges x 8 partials
        float s = 0.f;
#pragma unroll
        for (int j = 0; j < 16; ++j) s += tb[m][p * 16 + j] * cw2[p * 16 + j];
        parts[m][p] = s;
    }
    __syncthreads();
    if (tid < 32) {
        float coef = 0.f;
#pragma unroll
        for (int p = 0; p < 8; ++p) coef += parts[tid][p];
#pragma unroll
        for (int d = 0; d < 3; ++d)
            atomicAdd(&x_out[(size_t)rows[tid] * 3 + d], cdiff[tid][d] * coef);
    }
}

__global__ void __launch_bounds__(256) node_kernel(
    const float* __restrict__ h, const float* __restrict__ agg_h,
    const float* __restrict__ nb1, const float* __restrict__ nb2,
    const unsigned short* __restrict__ nw1s, const unsigned short* __restrict__ nw2s,
    float* __restrict__ h_out)
{
    __shared__ unsigned short nin[16][256];
    __shared__ unsigned short h1[16][128];

    int tid = threadIdx.x;
    int nbase = blockIdx.x * 16;

    for (int i = tid; i < 16 * 256; i += 256) {
        int m = i >> 8, k = i & 255;
        size_t node = nbase + m;
        float v = (k < 128) ? h[node * 128 + k] : agg_h[node * 128 + (k - 128)];
        nin[m][k] = f2bf(v);
    }
    __syncthreads();

    int wv = tid >> 5, lane = tid & 31;
    int ncol = wv * 16 + (lane & 15);
    int mrow0 = (lane >> 4) * 8;

    v8f acc = vzero();
#pragma unroll
    for (int kc = 0; kc < 8; ++kc) {
        v8u a = load_a(&nin[0][0], 256, kc, lane);
        v8u b = *(const v8u*)(nw1s + (((wv * 8 + kc) * 32 + lane) << 4));
        acc = wmma_bf16(a, b, acc);
    }
    {
        float bias = nb1[ncol];
#pragma unroll
        for (int r2 = 0; r2 < 8; ++r2) {
            int m = mrow0 + r2;
            h1[m][ncol] = f2bf(silu(acc[r2] + bias));
        }
    }
    __syncthreads();

    v8f acc2 = vzero();
#pragma unroll
    for (int kc = 0; kc < 4; ++kc) {
        v8u a = load_a(&h1[0][0], 128, kc, lane);
        v8u b = *(const v8u*)(nw2s + (((wv * 4 + kc) * 32 + lane) << 4));
        acc2 = wmma_bf16(a, b, acc2);
    }
    {
        float b2 = nb2[ncol];
#pragma unroll
        for (int r2 = 0; r2 < 8; ++r2) {
            size_t node = nbase + (mrow0 + r2);
            h_out[node * 128 + ncol] = h[node * 128 + ncol] + acc2[r2] + b2;
        }
    }
}

extern "C" void kernel_launch(void* const* d_in, const int* in_sizes, int n_in,
                              void* d_out, int out_size, void* d_ws, size_t ws_size,
                              hipStream_t stream) {
    const float* h   = (const float*)d_in[0];
    const float* x   = (const float*)d_in[1];
    const int*   ei  = (const int*)d_in[2];
    const float* ew1 = (const float*)d_in[3];
    const float* eb1 = (const float*)d_in[4];
    const float* ew2 = (const float*)d_in[5];
    const float* eb2 = (const float*)d_in[6];
    const float* cw1 = (const float*)d_in[7];
    const float* cb1 = (const float*)d_in[8];
    const float* cw2 = (const float*)d_in[9];
    const float* nw1 = (const float*)d_in[10];
    const float* nb1 = (const float*)d_in[11];
    const float* nw2 = (const float*)d_in[12];
    const float* nb2 = (const float*)d_in[13];

    int E = in_sizes[2] / 2;

    float* h_out = (float*)d_out;
    float* x_out = h_out + (size_t)NNODES * 128;

    char* ws = (char*)d_ws;
    unsigned short* we1s = (unsigned short*)(ws);            // 32768 bf16
    unsigned short* we2s = (unsigned short*)(ws + 65536);    // 16384 bf16
    unsigned short* wc1s = (unsigned short*)(ws + 98304);    // 16384 bf16
    unsigned short* nw1s = (unsigned short*)(ws + 131072);   // 32768 bf16
    unsigned short* nw2s = (unsigned short*)(ws + 196608);   // 16384 bf16
    float* agg_h = (float*)(ws + 229376);                    // 50000*128 fp32

    swizzle_w<<<32768 / 256, 256, 0, stream>>>(ew1, we1s, 8); // first 256 rows only
    swizzle_w<<<16384 / 256, 256, 0, stream>>>(ew2, we2s, 4);
    swizzle_w<<<16384 / 256, 256, 0, stream>>>(cw1, wc1s, 4);
    swizzle_w<<<32768 / 256, 256, 0, stream>>>(nw1, nw1s, 8);
    swizzle_w<<<16384 / 256, 256, 0, stream>>>(nw2, nw2s, 4);

    init_kernel<<<(NNODES * 128 + 255) / 256, 256, 0, stream>>>(x, agg_h, x_out);

    edge_kernel<<<E / 32, 256, 0, stream>>>(h, x, ei, E, ew1, eb1, eb2, cb1, cw2,
                                            we1s, we2s, wc1s, agg_h, x_out);

    node_kernel<<<NNODES / 16, 256, 0, stream>>>(h, agg_h, nb1, nb2, nw1s, nw2s, h_out);
}
